// GraphSwinWindowContext_1073741824681
// MI455X (gfx1250) — compile-verified, matrix operand order checked by reading, not run on hardware
//
#include <hip/hip_runtime.h>

// Problem constants (from reference)
#define Bn   32
#define Hn   192
#define Wn   192
#define Cn   64
#define NHn  32
#define NWn  32
#define NWIN (Bn * NHn * NWn)   // 32768 windows per partition
#define PITCH 68                // LDS row pitch in floats (bank-conflict pad)
#define WAVES 8                 // waves per block (256 threads)

typedef __attribute__((ext_vector_type(16))) __bf16 v16bf;
typedef __attribute__((ext_vector_type(8)))  float  v8f;

// Native f32 -> bf16 (RNE): should lower to v_cvt_{pk_}bf16_f32 on gfx1250,
// replacing the 3-4 VALU-op bit-twiddle per element.
__device__ __forceinline__ __bf16 to_bf16(float f) {
  return (__bf16)f;
}

// A fragment (16x32 bf16): lane half h owns K octets {kb..kb+7, kb+16..kb+23},
// kb = 32*kt + 8*h, per CDNA5 ISA 16-bit A-matrix table. Row = lane&15.
__device__ __forceinline__ v16bf afrag_lds(const float* row, int kb) {
  v16bf a;
#pragma unroll
  for (int j = 0; j < 8; ++j) a[j] = to_bf16(row[kb + j]);
#pragma unroll
  for (int j = 0; j < 8; ++j) a[j + 8] = to_bf16(row[kb + 16 + j]);
  return a;
}

// B fragment (32x16 bf16): mirrored K mapping, column n fixed per lane.
// Wm is row-major [64][64]: Wm[k*64 + n].
__device__ __forceinline__ v16bf bfrag_gmem(const float* Wm, int kb, int n) {
  v16bf b;
#pragma unroll
  for (int j = 0; j < 8; ++j) b[j] = to_bf16(Wm[(kb + j) * 64 + n]);
#pragma unroll
  for (int j = 0; j < 8; ++j) b[j + 8] = to_bf16(Wm[(kb + 16 + j) * 64 + n]);
  return b;
}

// One "block" layer on a 16x64 tile resident in per-wave LDS:
//   y = gelu(LN(x @ Wm + bias) * gamma + beta), written back into the tile.
// Safe to reuse the same tile: within one wave all fragment reads issue
// (in program order, DScnt-tracked) before any of the stores.
__device__ __forceinline__ void layer16(float* tile, const float* Wm,
                                        const float* bias, const float* gamma,
                                        const float* beta) {
  const int lane = threadIdx.x & 31;
  const int l15  = lane & 15;
  const int hlf  = lane >> 4;
  const int kb0  = hlf * 8;
  const int kb1  = 32 + hlf * 8;
  const float* row = tile + l15 * PITCH;

  v16bf a0 = afrag_lds(row, kb0);
  v16bf a1 = afrag_lds(row, kb1);

  v8f acc[4];
  float bn[4], gn[4], en[4];
#pragma unroll
  for (int nt = 0; nt < 4; ++nt) {
    const int n = nt * 16 + l15;
    v16bf b0 = bfrag_gmem(Wm, kb0, n);
    v16bf b1 = bfrag_gmem(Wm, kb1, n);
    v8f c = {};
    c = __builtin_amdgcn_wmma_f32_16x16x32_bf16(false, a0, false, b0,
                                                (short)0, c, false, false);
    c = __builtin_amdgcn_wmma_f32_16x16x32_bf16(false, a1, false, b1,
                                                (short)0, c, false, false);
    acc[nt] = c;
    bn[nt] = bias[n]; gn[nt] = gamma[n]; en[nt] = beta[n];
  }

  // C/D layout: VGPR v holds row (v + 8*hlf), channel nt*16 + l15.
  // LN over 64 channels = 4 local values + xor-reduce across 16 lanes.
#pragma unroll
  for (int v = 0; v < 8; ++v) {
    float s = 0.f, s2 = 0.f;
#pragma unroll
    for (int nt = 0; nt < 4; ++nt) {
      float x = acc[nt][v] + bn[nt];
      acc[nt][v] = x;
      s += x; s2 += x * x;
    }
#pragma unroll
    for (int m = 1; m < 16; m <<= 1) {
      s  += __shfl_xor(s,  m, 32);
      s2 += __shfl_xor(s2, m, 32);
    }
    const float mean = s * (1.f / 64.f);
    const float var  = s2 * (1.f / 64.f) - mean * mean;
    const float inv  = rsqrtf(var + 1e-5f);
    const int rowm = v + hlf * 8;
#pragma unroll
    for (int nt = 0; nt < 4; ++nt) {
      float x = (acc[nt][v] - mean) * inv * gn[nt] + en[nt];
      x = 0.5f * x * (1.f + erff(x * 0.70710678118654752f));   // exact-erf GELU
      tile[rowm * PITCH + nt * 16 + l15] = x;
    }
  }
}

// ---------------------------------------------------------------------------
// Kernel 1: 6x6 window mean-pool, regular + shifted partitions.
// grid = Bn*NHn*NWn blocks, 64 threads (one per channel, coalesced).
__global__ void pool_kernel(const float* __restrict__ x, float* __restrict__ pooled) {
  const int c   = threadIdx.x;
  const int blk = blockIdx.x;
  const int b   = blk >> 10;
  const int w   = blk & 1023;
  const int wy  = w >> 5, wx = w & 31;
  const float* base = x + (size_t)b * (Hn * Wn) * Cn;
  const int y0 = wy * 6, x0 = wx * 6;
  float sr = 0.f, ss = 0.f;
#pragma unroll
  for (int dy = 0; dy < 6; ++dy) {
    const int yr = y0 + dy;
    int ys = y0 + 3 + dy; if (ys >= Hn) ys -= Hn;   // shifted window rows
#pragma unroll
    for (int dx = 0; dx < 6; ++dx) {
      const int xr = x0 + dx;
      int xs = x0 + 3 + dx; if (xs >= Wn) xs -= Wn;
      sr += base[(yr * Wn + xr) * Cn + c];
      ss += base[(ys * Wn + xs) * Cn + c];
    }
  }
  pooled[(size_t)blk * Cn + c]          = sr * (1.f / 36.f);
  pooled[(size_t)(NWIN + blk) * Cn + c] = ss * (1.f / 36.f);
}

// ---------------------------------------------------------------------------
// Kernel 2: token block over 2*NWIN pooled rows (linear GEMM+LN+GELU).
// 256 threads = 8 waves, one 16-row tile per wave. 512 blocks * 8 = 4096 tiles.
__global__ void token_kernel(const float* __restrict__ pooled, float* __restrict__ tok,
                             const float* __restrict__ Wm, const float* __restrict__ bias,
                             const float* __restrict__ gamma, const float* __restrict__ beta) {
  __shared__ float tiles[WAVES][16 * PITCH];
  const int wid  = threadIdx.x >> 5;
  const int lane = threadIdx.x & 31;
  const int l15  = lane & 15, hlf = lane >> 4;
  const int tileIdx = blockIdx.x * WAVES + wid;
  float* tile = tiles[wid];

  // stage 16 rows x 64 ch: lane handles row l15, half of channels
  const float* src = pooled + ((size_t)tileIdx * 16 + l15) * Cn + hlf * 32;
  float* dst = tile + l15 * PITCH + hlf * 32;
#pragma unroll
  for (int j = 0; j < 8; ++j) ((float4*)dst)[j] = ((const float4*)src)[j];

  layer16(tile, Wm, bias, gamma, beta);

  float* out = tok + ((size_t)tileIdx * 16 + l15) * Cn + hlf * 32;
#pragma unroll
  for (int j = 0; j < 8; ++j) ((float4*)out)[j] = ((const float4*)dst)[j];
}

// ---------------------------------------------------------------------------
// Kernel 3: per-quadrant fuse+out blocks, then broadcast each row to its
// 3x3 pixel patch. Rows r = ((b*1024 + wi)*4 + q), 131072 total; 8192 tiles.
__global__ void fuse_out_kernel(const float* __restrict__ tok, float* __restrict__ out,
                                const float* __restrict__ fW, const float* __restrict__ fb,
                                const float* __restrict__ fg, const float* __restrict__ fe,
                                const float* __restrict__ oW, const float* __restrict__ ob,
                                const float* __restrict__ og, const float* __restrict__ oe) {
  __shared__ float tiles[WAVES][16 * PITCH];
  const int wid  = threadIdx.x >> 5;
  const int lane = threadIdx.x & 31;
  const int l15  = lane & 15, hlf = lane >> 4;
  const int tileIdx = blockIdx.x * WAVES + wid;
  float* tile = tiles[wid];

  // gather: row = tok_reg[b,wy,wx] + tok_shift[b,(wy+qy-1)%32,(wx+qx-1)%32]
  {
    const int r  = tileIdx * 16 + l15;
    const int q  = r & 3, wi = (r >> 2) & 1023, b = r >> 12;
    const int wy = wi >> 5, wx = wi & 31;
    const int qy = q >> 1,  qx = q & 1;
    const int sy = (wy + qy + 31) & 31;
    const int sx = (wx + qx + 31) & 31;
    const float* tr = tok + ((size_t)(b * 1024 + wi)) * Cn + hlf * 32;
    const float* ts = tok + ((size_t)(NWIN + b * 1024 + sy * 32 + sx)) * Cn + hlf * 32;
    float* dst = tile + l15 * PITCH + hlf * 32;
#pragma unroll
    for (int j = 0; j < 32; j += 4) {
      float4 a = *(const float4*)(tr + j);
      float4 c = *(const float4*)(ts + j);
      a.x += c.x; a.y += c.y; a.z += c.z; a.w += c.w;
      *(float4*)(dst + j) = a;
    }
  }

  layer16(tile, fW, fb, fg, fe);   // fuse block
  layer16(tile, oW, ob, og, oe);   // out block

  // broadcast each of the 16 rows to its 3x3 pixel quadrant (float2/lane)
#pragma unroll 1
  for (int m = 0; m < 16; ++m) {
    const int r  = tileIdx * 16 + m;
    const int q  = r & 3, wi = (r >> 2) & 1023, b = r >> 12;
    const int wy = wi >> 5, wx = wi & 31;
    const int y0 = wy * 6 + (q >> 1) * 3;
    const int x0 = wx * 6 + (q & 1) * 3;
    const float2 val = *(const float2*)(tile + m * PITCH + lane * 2);
    float* obase = out + (size_t)b * (Hn * Wn) * Cn;
#pragma unroll
    for (int dy = 0; dy < 3; ++dy)
#pragma unroll
      for (int dx = 0; dx < 3; ++dx) {
        *(float2*)(obase + ((y0 + dy) * Wn + (x0 + dx)) * Cn + lane * 2) = val;
      }
  }
}

// ---------------------------------------------------------------------------
extern "C" void kernel_launch(void* const* d_in, const int* in_sizes, int n_in,
                              void* d_out, int out_size, void* d_ws, size_t ws_size,
                              hipStream_t stream) {
  const float* h_pixel = (const float*)d_in[0];
  const float* token_W = (const float*)d_in[1];
  const float* token_b = (const float*)d_in[2];
  const float* token_g = (const float*)d_in[3];
  const float* token_e = (const float*)d_in[4];
  const float* fuse_W  = (const float*)d_in[5];
  const float* fuse_b  = (const float*)d_in[6];
  const float* fuse_g  = (const float*)d_in[7];
  const float* fuse_e  = (const float*)d_in[8];
  const float* out_W   = (const float*)d_in[9];
  const float* out_b   = (const float*)d_in[10];
  const float* out_g   = (const float*)d_in[11];
  const float* out_e   = (const float*)d_in[12];

  float* ws     = (float*)d_ws;
  float* pooled = ws;                         // [2*NWIN][64]
  float* tok    = ws + (size_t)2 * NWIN * Cn; // [2*NWIN][64]

  pool_kernel<<<NWIN, 64, 0, stream>>>(h_pixel, pooled);
  token_kernel<<<(2 * NWIN / 16) / WAVES, 32 * WAVES, 0, stream>>>(
      pooled, tok, token_W, token_b, token_g, token_e);
  fuse_out_kernel<<<(4 * NWIN / 16) / WAVES, 32 * WAVES, 0, stream>>>(
      tok, (float*)d_out, fuse_W, fuse_b, fuse_g, fuse_e,
      out_W, out_b, out_g, out_e);
}
